// WLSDiscriminatorLinear_49065706389955
// MI455X (gfx1250) — matrix operand order repaired and checked by reading, not compiled, not run on hardware
//
#include <hip/hip_runtime.h>
#include <hip/hip_bf16.h>

// Problem constants (fixed by reference)
#define BATCH 256
#define NNODE 128
#define EBCH  5
#define NATT  64
#define DIMH  512
#define NITER 3

typedef __attribute__((ext_vector_type(16))) __bf16 v16bf;
typedef __attribute__((ext_vector_type(8)))  float  v8f;

union FragU { v16bf v; uint4 q[2]; };

// ---------- prep kernels ----------

// Ared[b,n,m] = sum_{e=1..4} adj[b,n,m,e]  (channel 0 dropped), stored bf16
__global__ void reduce_adj_k(const float* __restrict__ adj, __bf16* __restrict__ out) {
  long i = (long)blockIdx.x * blockDim.x + threadIdx.x;   // over B*N*N
  const float* p = adj + i * EBCH;
  float s = p[1] + p[2] + p[3] + p[4];
  out[i] = (__bf16)s;
}

__global__ void cvt_bf16_k(const float* __restrict__ in, __bf16* __restrict__ out, long n) {
  long i = (long)blockIdx.x * blockDim.x + threadIdx.x;
  if (i < n) out[i] = (__bf16)in[i];
}

// WT[n,k] = (bf16) W[k,n] ;  W row length = Nc, WT row length = K
__global__ void transpose_w_k(const float* __restrict__ W, __bf16* __restrict__ WT,
                              int K, int Nc, long total) {
  long i = (long)blockIdx.x * blockDim.x + threadIdx.x;
  if (i >= total) return;
  long n = i / K, k = i % K;
  WT[n * (long)K + k] = (__bf16)W[k * (long)Nc + n];
}

// ---------- generic WMMA GEMM: C[b] = op( A[b](128xK) * Bt[b](NcolsxK)^T ) ----------
// A row-major [128 x K]; Bt row-major [Ncols x K] (i.e. B transposed).
// B panel (64 cols x K) is staged in LDS once per column group and shared by
// all 8 waves (ds_load_b128 fragment reads, +8 bf16 row pad -> conflict-free).
// Epilogue: +bias (opt), +addsrc[b] row-major (opt), relu (opt),
//           store C row-major bf16, optional transposed store Ct [Ncols x 128].
__global__ __launch_bounds__(256)
void wmma_gemm_k(const __bf16* __restrict__ A, long sA,
                 const __bf16* __restrict__ Bt, long sB,
                 const float* __restrict__ bias,
                 const __bf16* __restrict__ addsrc, long sAdd,
                 __bf16* __restrict__ C, long sC,
                 __bf16* __restrict__ Ct, long sCt,
                 int Ncols, int K, int relu) {
  __shared__ alignas(16) __bf16 sBp[64 * (DIMH + 8)];   // 65 KB max (K=512)

  const int b    = blockIdx.x;
  const int tid  = threadIdx.x;
  const int wave = tid >> 5;          // 8 waves, 16 rows each -> M=128
  const int lane = tid & 31;
  const int row16 = lane & 15;
  const int sel   = lane >> 4;        // lane half select
  const int m0    = wave * 16;
  const int ldl   = K + 8;            // padded LDS row stride (bf16 elems)

  const __bf16* Ab = A  + (long)b * sA;
  const __bf16* Bb = Bt + (long)b * sB;

  for (int n0 = 0; n0 < Ncols; n0 += 64) {
    // ---- cooperative stage of B panel [n0..n0+64) x K into LDS ----
    {
      const int chunksPerRow = K >> 3;          // 16B chunks per row
      const int totalChunks  = 64 * chunksPerRow;
      for (int c = tid; c < totalChunks; c += 256) {
        const int rr = c / chunksPerRow;
        const int kk = (c - rr * chunksPerRow) << 3;
        *reinterpret_cast<uint4*>(&sBp[rr * ldl + kk]) =
            *reinterpret_cast<const uint4*>(Bb + (long)(n0 + rr) * K + kk);
      }
    }
    __syncthreads();

    v8f acc[4] = {};                  // 4 column tiles of 16x16 f32
    for (int k0 = 0; k0 < K; k0 += 32) {
      // A fragment: row = m0+row16 ; halves 0..7 -> K = k0+8*sel+0..7,
      //             halves 8..15 -> K = k0+16+8*sel+0..7  (ISA 16-bit A layout)
      FragU afr;
      const __bf16* ap = Ab + (long)(m0 + row16) * K + k0 + sel * 8;
      afr.q[0] = *reinterpret_cast<const uint4*>(ap);
      afr.q[1] = *reinterpret_cast<const uint4*>(ap + 16);
#pragma unroll
      for (int t = 0; t < 4; ++t) {
        // B fragment from LDS: col row16 of tile t ; same K layout as A fragment
        FragU bfr;
        const __bf16* bp = &sBp[(t * 16 + row16) * ldl + k0 + sel * 8];
        bfr.q[0] = *reinterpret_cast<const uint4*>(bp);
        bfr.q[1] = *reinterpret_cast<const uint4*>(bp + 16);
        acc[t] = __builtin_amdgcn_wmma_f32_16x16x32_bf16(
            false, afr.v, false, bfr.v, (short)0, acc[t], false, false);
      }
    }

    // epilogue: C layout lane -> N = n0+16t+row16 ; vgpr r -> M = m0 + 8*sel + r
#pragma unroll
    for (int t = 0; t < 4; ++t) {
      const int col = n0 + t * 16 + row16;
      const float bv = bias ? bias[col] : 0.0f;
#pragma unroll
      for (int r = 0; r < 8; ++r) {
        const int mrow = m0 + sel * 8 + r;
        float v = acc[t][r] + bv;
        if (addsrc) v += (float)addsrc[(long)b * sAdd + (long)mrow * Ncols + col];
        if (relu && v < 0.0f) v = 0.0f;
        const __bf16 o = (__bf16)v;
        C[(long)b * sC + (long)mrow * Ncols + col] = o;
        if (Ct) Ct[(long)b * sCt + (long)col * NNODE + mrow] = o;
      }
    }
    __syncthreads();   // panel reuse barrier before restaging
  }
}

// ---------- finalize: graph_repr = mean_n h ; logit = graph_repr @ W_out + b_out ----------
__global__ __launch_bounds__(256)
void finalize_k(const __bf16* __restrict__ h, const float* __restrict__ Wout,
                const float* __restrict__ bout, float* __restrict__ out) {
  const int b = blockIdx.x, tid = threadIdx.x;
  __shared__ float red[256];
  float partial = 0.0f;
  for (int d = tid; d < DIMH; d += 256) {
    float s = 0.0f;
    const __bf16* col = h + ((long)b * NNODE) * DIMH + d;
    for (int n = 0; n < NNODE; ++n) s += (float)col[(long)n * DIMH];
    const float gr = s * (1.0f / NNODE);
    out[BATCH + (long)b * DIMH + d] = gr;   // graph_repr after logits
    partial += gr * Wout[d];
  }
  red[tid] = partial;
  __syncthreads();
  for (int s = 128; s > 0; s >>= 1) {
    if (tid < s) red[tid] += red[tid + s];
    __syncthreads();
  }
  if (tid == 0) out[b] = red[0] + bout[0];
}

// ---------- host side ----------
extern "C" void kernel_launch(void* const* d_in, const int* in_sizes, int n_in,
                              void* d_out, int out_size, void* d_ws, size_t ws_size,
                              hipStream_t stream) {
  const float* adj      = (const float*)d_in[0];
  // d_in[1] = hidden (unused by forward)
  const float* node     = (const float*)d_in[2];
  const float* W_embed  = (const float*)d_in[3];
  const float* b_embed  = (const float*)d_in[4];
  const float* W_layers = (const float*)d_in[5];
  const float* b_layers = (const float*)d_in[6];
  const float* W_out    = (const float*)d_in[7];
  const float* b_out    = (const float*)d_in[8];
  float* out = (float*)d_out;

  char* ws = (char*)d_ws;
  const size_t SZ_ARED = (size_t)BATCH * NNODE * NNODE * 2;       // 8.4 MB
  const size_t SZ_NODE = (size_t)BATCH * NNODE * NATT * 2;        // 4.2 MB
  const size_t SZ_WTE  = (size_t)DIMH * NATT * 2;                 // 64 KB
  const size_t SZ_WTL  = (size_t)NITER * DIMH * DIMH * 2;         // 1.5 MB
  const size_t SZ_H    = (size_t)BATCH * NNODE * DIMH * 2;        // 33.6 MB
  __bf16* Ared   = (__bf16*)(ws);
  __bf16* nodeBF = (__bf16*)(ws + SZ_ARED);
  __bf16* WTe    = (__bf16*)(ws + SZ_ARED + SZ_NODE);
  __bf16* WTl    = (__bf16*)(ws + SZ_ARED + SZ_NODE + SZ_WTE);
  __bf16* hRM    = (__bf16*)(ws + SZ_ARED + SZ_NODE + SZ_WTE + SZ_WTL);
  __bf16* hT     = (__bf16*)(ws + SZ_ARED + SZ_NODE + SZ_WTE + SZ_WTL + SZ_H);
  __bf16* tmpRM  = (__bf16*)(ws + SZ_ARED + SZ_NODE + SZ_WTE + SZ_WTL + 2 * SZ_H);

  // --- prep ---
  reduce_adj_k<<<(BATCH * NNODE * NNODE) / 256, 256, 0, stream>>>(adj, Ared);
  cvt_bf16_k<<<(BATCH * NNODE * NATT) / 256, 256, 0, stream>>>(node, nodeBF,
                                                               (long)BATCH * NNODE * NATT);
  transpose_w_k<<<(DIMH * NATT + 255) / 256, 256, 0, stream>>>(W_embed, WTe, NATT, DIMH,
                                                               (long)DIMH * NATT);
  for (int l = 0; l < NITER; ++l)
    transpose_w_k<<<(DIMH * DIMH + 255) / 256, 256, 0, stream>>>(
        W_layers + (long)l * DIMH * DIMH, WTl + (long)l * DIMH * DIMH, DIMH, DIMH,
        (long)DIMH * DIMH);

  // --- embed: h = node @ W_embed + b_embed ---
  wmma_gemm_k<<<BATCH, 256, 0, stream>>>(
      nodeBF, (long)NNODE * NATT, WTe, 0, b_embed,
      nullptr, 0, hRM, (long)NNODE * DIMH, hT, (long)NNODE * DIMH,
      DIMH, NATT, 0);

  // --- 3 WLS layers ---
  for (int i = 0; i < NITER; ++i) {
    // tmp = h + Ared @ h
    wmma_gemm_k<<<BATCH, 256, 0, stream>>>(
        Ared, (long)NNODE * NNODE, hT, (long)NNODE * DIMH, nullptr,
        hRM, (long)NNODE * DIMH, tmpRM, (long)NNODE * DIMH, nullptr, 0,
        DIMH, NNODE, 0);
    // h = relu(tmp @ W_i + b_i)
    wmma_gemm_k<<<BATCH, 256, 0, stream>>>(
        tmpRM, (long)NNODE * DIMH, WTl + (long)i * DIMH * DIMH, 0,
        b_layers + (long)i * DIMH, nullptr, 0,
        hRM, (long)NNODE * DIMH, hT, (long)NNODE * DIMH,
        DIMH, DIMH, 1);
  }

  // --- mean over nodes + output head ---
  finalize_k<<<BATCH, 256, 0, stream>>>(hRM, W_out, b_out, out);
}
